// SoftTreeClassifier_16630113370386
// MI455X (gfx1250) — matrix operand order repaired and checked by reading, not compiled, not run on hardware
//
#include <hip/hip_runtime.h>
#include <hip/hip_bf16.h>

// ---------------- types for WMMA ----------------
typedef __attribute__((ext_vector_type(16))) __bf16          v16bf;
typedef __attribute__((ext_vector_type(16))) unsigned short  v16us;
typedef __attribute__((ext_vector_type(8)))  unsigned int    v8u;
typedef __attribute__((ext_vector_type(8)))  float           v8f;

#define DEPTH          7
#define INPUT_DIM      128
#define N_CLASSES      64
#define N_INTERNAL     127
#define N_LEAVES       128
#define ROWS_PER_WAVE  16
#define WAVES_PER_BLK  8
#define ROWS_PER_BLK   (ROWS_PER_WAVE * WAVES_PER_BLK)   // 128
#define S_STRIDE       132                               // 128 + 4 pad: rows hit distinct LDS banks

// one-time (preprocessing) RNE float->bf16
__device__ __forceinline__ unsigned short f2bf(float f) {
    unsigned u = __builtin_bit_cast(unsigned, f);
    u += 0x7FFFu + ((u >> 16) & 1u);
    return (unsigned short)(u >> 16);
}

// hot path: pack two floats into a bf16 pair with ONE v_perm_b32 (truncate)
__device__ __forceinline__ unsigned pk2bf(float lo, float hi) {
    return __builtin_amdgcn_perm(__builtin_bit_cast(unsigned, hi),
                                 __builtin_bit_cast(unsigned, lo),
                                 0x07060302u);
}

__global__ __launch_bounds__(256, 1)
void soft_tree_kernel(const float* __restrict__ X,
                      const float* __restrict__ W,
                      const float* __restrict__ b,
                      const float* __restrict__ leaf_scores,
                      float* __restrict__ out,
                      int numTiles)
{
    __shared__ unsigned short Wl[INPUT_DIM * 128];            // W^T bf16 [k][node], 32 KB
    __shared__ unsigned short Ql[N_LEAVES * N_CLASSES];       // softmax(leaf) bf16 [leaf][class], 16 KB
    __shared__ float          bl[128];
    __shared__ float          Sl[WAVES_PER_BLK * ROWS_PER_WAVE * S_STRIDE]; // 66 KB

    const int tid = threadIdx.x;

    // ---------------- one-time block preprocessing ----------------
    for (int i = tid; i < INPUT_DIM * 128; i += 256) {
        int k = i >> 7, n = i & 127;
        float v = (n < N_INTERNAL) ? W[n * INPUT_DIM + k] : 0.0f;
        Wl[k * 128 + n] = f2bf(v);
    }
    if (tid < N_LEAVES) {
        const float* row = leaf_scores + tid * N_CLASSES;
        float mx = row[0];
        #pragma unroll
        for (int c = 1; c < N_CLASSES; ++c) mx = fmaxf(mx, row[c]);
        float sum = 0.0f;
        #pragma unroll
        for (int c = 0; c < N_CLASSES; ++c) sum += __expf(row[c] - mx);
        float inv = 1.0f / sum;
        #pragma unroll
        for (int c = 0; c < N_CLASSES; ++c)
            Ql[tid * N_CLASSES + c] = f2bf(__expf(row[c] - mx) * inv);
        bl[tid] = (tid < N_INTERNAL) ? b[tid] : 0.0f;
    }
    __syncthreads();

    const int lane  = tid & 31;
    const int wave  = tid >> 5;
    const int rloc  = lane & 15;        // A-frag row / C-frag column index
    const int khalf = lane >> 4;        // which K-half this lane holds (A layout)
    float* Srow = &Sl[wave * ROWS_PER_WAVE * S_STRIDE];
    const int rowOff = rloc * S_STRIDE; // base into Srow for the mu stage

    // ---------------- grid-stride over 128-row tiles ----------------
    for (int tile = blockIdx.x; tile < numTiles; tile += gridDim.x) {
        const int rowBase = tile * ROWS_PER_BLK + wave * ROWS_PER_WAVE;
        const float* xrow = X + (size_t)(rowBase + rloc) * INPUT_DIM;

        if (tile + (int)gridDim.x < numTiles)
            __builtin_prefetch(xrow + (size_t)gridDim.x * ROWS_PER_BLK * INPUT_DIM, 0, 1);

        // ---- load A fragments of X (16 rows x 128 K), f32 -> bf16 via v_perm ----
        v16bf aF[4];
        #pragma unroll
        for (int kb4 = 0; kb4 < 4; ++kb4) {
            const float* b0 = xrow + kb4 * 32 + khalf * 8;
            float4 f0 = *(const float4*)(b0);
            float4 f1 = *(const float4*)(b0 + 4);
            float4 f2 = *(const float4*)(b0 + 16);
            float4 f3 = *(const float4*)(b0 + 20);
            v8u u;
            u[0] = pk2bf(f0.x, f0.y); u[1] = pk2bf(f0.z, f0.w);
            u[2] = pk2bf(f1.x, f1.y); u[3] = pk2bf(f1.z, f1.w);
            u[4] = pk2bf(f2.x, f2.y); u[5] = pk2bf(f2.z, f2.w);
            u[6] = pk2bf(f3.x, f3.y); u[7] = pk2bf(f3.z, f3.w);
            aF[kb4] = __builtin_bit_cast(v16bf, u);
        }

        // ---- GEMM1: S = sigmoid(X Wt + b), 8 node-tiles of 16 ----
        #pragma unroll
        for (int nt = 0; nt < 8; ++nt) {
            v8f acc = {};
            #pragma unroll
            for (int kb4 = 0; kb4 < 4; ++kb4) {
                v16us bU;
                #pragma unroll
                for (int s = 0; s < 16; ++s)
                    bU[s] = Wl[(kb4 * 32 + khalf * 8 + (s & 7) + 16 * (s >> 3)) * 128 + nt * 16 + rloc];
                v16bf bF = __builtin_bit_cast(v16bf, bU);
                acc = __builtin_amdgcn_wmma_f32_16x16x32_bf16(
                          false, aF[kb4], false, bF, (short)0, acc, false, false);
            }
            float bv = bl[nt * 16 + rloc];
            #pragma unroll
            for (int r = 0; r < 8; ++r) {
                float s = 1.0f / (1.0f + __expf(-(acc[r] + bv)));
                Srow[(khalf * 8 + r) * S_STRIDE + nt * 16 + rloc] = s;
            }
        }
        // cross-lane LDS RAW inside the wave: wait for DS stores to land
        asm volatile("s_wait_dscnt 0" ::: "memory");

        // ---- mu path products: explicit shared-prefix tree per 8-leaf block ----
        // leaf bits (MSB..LSB): b6 b5 b4 b3 b2 b1 b0 ; block base C has b2..b0 = 0
        v16bf muF[4];
        #pragma unroll
        for (int kb4 = 0; kb4 < 4; ++kb4) {
            v8u u;
            #pragma unroll
            for (int half = 0; half < 2; ++half) {
                const int C = kb4 * 32 + half * 16 + khalf * 8;  // base leaf of 8-block
                // levels 0..3: constant per block
                float P = 1.0f;
                #pragma unroll
                for (int d = 0; d < 4; ++d) {
                    int node = (1 << d) - 1 + (C >> (DEPTH - d));
                    float p  = Srow[rowOff + node];
                    int bit  = (C >> (DEPTH - 1 - d)) & 1;  // runtime only at d==3
                    P *= bit ? (1.0f - p) : p;
                }
                // level 4 (bit2): one node
                float p4 = Srow[rowOff + 15 + (C >> 3)];
                float m0 = P * p4;
                float m1 = P * (1.0f - p4);
                // level 5 (bit1): two nodes
                float p5a = Srow[rowOff + 31 + (C >> 2)];
                float p5b = Srow[rowOff + 31 + (C >> 2) + 1];
                float m00 = m0 * p5a, m01 = m0 * (1.0f - p5a);
                float m10 = m1 * p5b, m11 = m1 * (1.0f - p5b);
                // level 6 (bit0): four nodes
                float p60 = Srow[rowOff + 63 + (C >> 1)];
                float p61 = Srow[rowOff + 63 + (C >> 1) + 1];
                float p62 = Srow[rowOff + 63 + (C >> 1) + 2];
                float p63 = Srow[rowOff + 63 + (C >> 1) + 3];
                float mu0 = fmaxf(m00 * p60,          1e-12f);
                float mu1 = fmaxf(m00 * (1.0f - p60), 1e-12f);
                float mu2 = fmaxf(m01 * p61,          1e-12f);
                float mu3 = fmaxf(m01 * (1.0f - p61), 1e-12f);
                float mu4 = fmaxf(m10 * p62,          1e-12f);
                float mu5 = fmaxf(m10 * (1.0f - p62), 1e-12f);
                float mu6 = fmaxf(m11 * p63,          1e-12f);
                float mu7 = fmaxf(m11 * (1.0f - p63), 1e-12f);
                u[half * 4 + 0] = pk2bf(mu0, mu1);
                u[half * 4 + 1] = pk2bf(mu2, mu3);
                u[half * 4 + 2] = pk2bf(mu4, mu5);
                u[half * 4 + 3] = pk2bf(mu6, mu7);
            }
            muF[kb4] = __builtin_bit_cast(v16bf, u);
        }

        // ---- GEMM2: out = log(mu Q), 4 class-tiles of 16 ----
        #pragma unroll
        for (int ct = 0; ct < 4; ++ct) {
            v8f acc = {};
            #pragma unroll
            for (int kb4 = 0; kb4 < 4; ++kb4) {
                v16us qU;
                #pragma unroll
                for (int s = 0; s < 16; ++s)
                    qU[s] = Ql[(kb4 * 32 + khalf * 8 + (s & 7) + 16 * (s >> 3)) * N_CLASSES + ct * 16 + rloc];
                v16bf bF = __builtin_bit_cast(v16bf, qU);
                acc = __builtin_amdgcn_wmma_f32_16x16x32_bf16(
                          false, muF[kb4], false, bF, (short)0, acc, false, false);
            }
            #pragma unroll
            for (int r = 0; r < 8; ++r) {
                out[(size_t)(rowBase + khalf * 8 + r) * N_CLASSES + ct * 16 + rloc]
                    = __logf(acc[r]);
            }
        }
    }
}

extern "C" void kernel_launch(void* const* d_in, const int* in_sizes, int n_in,
                              void* d_out, int out_size, void* d_ws, size_t ws_size,
                              hipStream_t stream) {
    const float* X  = (const float*)d_in[0];
    const float* W  = (const float*)d_in[1];
    const float* b  = (const float*)d_in[2];
    const float* ls = (const float*)d_in[3];
    float* out = (float*)d_out;

    int batch    = in_sizes[0] / INPUT_DIM;          // 524288
    int numTiles = batch / ROWS_PER_BLK;             // 4096 full tiles
    int grid     = numTiles < 1024 ? (numTiles > 0 ? numTiles : 1) : 1024;

    soft_tree_kernel<<<grid, 256, 0, stream>>>(X, W, b, ls, out, numTiles);
}
